// ABR_32512902430754
// MI455X (gfx1250) — matrix operand order repaired and checked by reading, not compiled, not run on hardware
//
#include <hip/hip_runtime.h>
#include <hip/hip_bf16.h>
#include <math.h>
#include <stdint.h>

// ---------------------------------------------------------------------------
// DCNv2 x2 for MI455X (gfx1250, wave32).
// - GEMMs on v_wmma_f32_16x16x32_bf16, split-bf16 (hi/lo) 3-pass accumulation.
// - Activations kept in NHWC so every 16-channel K-slice is 64 contiguous
//   bytes -> b128 loads for both the dense conv and the bilinear gathers.
// - Per-tile sample records staged into LDS via the Tensor Data Mover.
// ---------------------------------------------------------------------------

typedef __bf16 bf16_t;
typedef __attribute__((ext_vector_type(16))) __bf16 v16bf;
typedef __attribute__((ext_vector_type(8)))  float  v8f;
typedef __attribute__((ext_vector_type(4)))  float  v4f;
typedef __attribute__((ext_vector_type(4)))  unsigned int v4u;
typedef __attribute__((ext_vector_type(4)))  int    v4i;
typedef __attribute__((ext_vector_type(8)))  int    v8i;

#define Bn 4
#define Cn 64
#define Hn 128
#define Wn 128
#define Nk 9
#define NCHUNK 18           // K = 576 = 18 chunks of 32

struct Rec { int idx[4]; float wt[4]; };   // 32 B: 4 gather indices + weights

union F16x { v4f f4[4]; float f[16]; };

__device__ __forceinline__ void split_bf16(float v, bf16_t& hi, bf16_t& lo) {
  hi = (bf16_t)v;
  lo = (bf16_t)(v - (float)hi);
}

// ---------------------------------------------------------------------------
// NCHW -> NHWC transpose (LDS-tiled, 64ch x 64w per block).
// ---------------------------------------------------------------------------
__global__ void __launch_bounds__(256)
nchw_to_nhwc_kernel(const float* __restrict__ src, float* __restrict__ dst) {
  __shared__ float tile[64 * 65];
  const int tid = threadIdx.x;
  const int w0 = (blockIdx.x & 1) * 64;
  const int h  = (blockIdx.x >> 1) & (Hn - 1);
  const int b  = blockIdx.x >> 8;
#pragma unroll
  for (int s = 0; s < 16; ++s) {
    int idx = s * 256 + tid;
    int c = idx >> 6, i = idx & 63;
    tile[c * 65 + i] = src[(((size_t)b * Cn + c) * Hn + h) * Wn + w0 + i];
  }
  __syncthreads();
#pragma unroll
  for (int s = 0; s < 16; ++s) {
    int idx = s * 256 + tid;
    int i = idx >> 6, c = idx & 63;
    dst[(((size_t)b * Hn + h) * Wn + w0 + i) * Cn + c] = tile[c * 65 + i];
  }
}

// ---------------------------------------------------------------------------
// Pack A matrix (M = MT*16, K = 576, k = p*64+ci) into per-lane WMMA frags.
// A k order used here (contiguous, B matches): element j of v16bf -> kk = (lane&16?16:0)+j.
// dst record ((t*MT+mt)*32+lane)*32 bf16: [0..15]=hi, [16..31]=lo.
// ---------------------------------------------------------------------------
__global__ void pack_a_kernel(const float* __restrict__ srcA, int MA,
                              const float* __restrict__ srcB, int MB,
                              bf16_t* __restrict__ dst, int MT) {
  int idx = blockIdx.x * blockDim.x + threadIdx.x;
  int total = NCHUNK * MT * 32 * 16;
  if (idx >= total) return;
  int j    = idx & 15;
  int lane = (idx >> 4) & 31;
  int mtt  = idx >> 9;
  int mt   = mtt % MT;
  int t    = mtt / MT;
  int m    = mt * 16 + (lane & 15);
  int kk   = ((lane & 16) ? 16 : 0) + j;
  int k    = t * 32 + kk;
  int ci   = k & 63;
  int p    = k >> 6;
  float v = 0.f;
  if (m < MA)           v = srcA[((size_t)m * Cn + ci) * Nk + p];
  else if (m < MA + MB) v = srcB[((size_t)(m - MA) * Cn + ci) * Nk + p];
  bf16_t hi, lo; split_bf16(v, hi, lo);
  bf16_t* d = dst + (((size_t)mtt * 32) + lane) * 32;
  d[j]      = hi;
  d[16 + j] = lo;
}

// ---------------------------------------------------------------------------
// Dense 3x3 conv (pad=1) -> 27 channels (18 offsets + 9 sigmoid masks).
// Implicit GEMM M=32, K=576; input NHWC so each lane's 16-channel slice is
// 4 x b128.  6 WMMAs per K-chunk.
// ---------------------------------------------------------------------------
__global__ void __launch_bounds__(128)
conv_offmask_kernel(const float* __restrict__ xT,       // NHWC
                    const bf16_t* __restrict__ apack,   // MT=2
                    const float* __restrict__ b_p,
                    const float* __restrict__ b_m,
                    float* __restrict__ offmask) {
  const int lane = threadIdx.x & 31;
  const int wv   = threadIdx.x >> 5;
  const int tile = blockIdx.x * 4 + wv;       // 4096 tiles, exact
  const int w0   = (tile & 7) * 16;
  const int h    = (tile >> 3) & (Hn - 1);
  const int b    = tile >> 10;
  const int px   = w0 + (lane & 15);
  const int ciL  = lane & 16;

  v8f acc0 = {}; v8f acc1 = {};
  const float* xb = xT + (size_t)b * Hn * Wn * Cn;

  for (int t = 0; t < NCHUNK; ++t) {
    const int p   = t >> 1;
    const int ci0 = (t & 1) * 32 + ciL;
    const int dy  = p / 3 - 1, dx = p % 3 - 1;
    const int row = h + dy;
    const int col = px + dx;
    const int rowc = min(max(row, 0), Hn - 1);
    const int colc = min(max(col, 0), Wn - 1);
    const float s = (row >= 0 && row < Hn && col >= 0 && col < Wn) ? 1.f : 0.f;

    F16x bx;
    const v4f* q = (const v4f*)(xb + (((size_t)rowc * Wn) + colc) * Cn + ci0);
    bx.f4[0] = q[0]; bx.f4[1] = q[1]; bx.f4[2] = q[2]; bx.f4[3] = q[3];

    union { v16bf v; bf16_t e[16]; } bh, bl;
#pragma unroll
    for (int j = 0; j < 16; ++j) split_bf16(s * bx.f[j], bh.e[j], bl.e[j]);

    const bf16_t* ap = apack + (((size_t)t * 2) * 32 + lane) * 32;
    v16bf a0h = *(const v16bf*)(ap);
    v16bf a0l = *(const v16bf*)(ap + 16);
    v16bf a1h = *(const v16bf*)(ap + 1024);
    v16bf a1l = *(const v16bf*)(ap + 1040);

    acc0 = __builtin_amdgcn_wmma_f32_16x16x32_bf16(false, a0h, false, bh.v, (short)0, acc0, false, false);
    acc0 = __builtin_amdgcn_wmma_f32_16x16x32_bf16(false, a0h, false, bl.v, (short)0, acc0, false, false);
    acc0 = __builtin_amdgcn_wmma_f32_16x16x32_bf16(false, a0l, false, bh.v, (short)0, acc0, false, false);
    acc1 = __builtin_amdgcn_wmma_f32_16x16x32_bf16(false, a1h, false, bh.v, (short)0, acc1, false, false);
    acc1 = __builtin_amdgcn_wmma_f32_16x16x32_bf16(false, a1h, false, bl.v, (short)0, acc1, false, false);
    acc1 = __builtin_amdgcn_wmma_f32_16x16x32_bf16(false, a1l, false, bh.v, (short)0, acc1, false, false);
  }

  const int rsel = (lane & 16) ? 8 : 0;
  float* om = offmask + (size_t)b * 27 * Hn * Wn;
#pragma unroll
  for (int r = 0; r < 8; ++r) {
    {
      int ch = r + rsel;          // 0..15: always offset channel
      om[((size_t)ch * Hn + h) * Wn + px] = acc0[r] + b_p[ch];
    }
    {
      int ch = 16 + r + rsel;     // 16..31
      float v = acc1[r];
      if (ch < 18)
        om[((size_t)ch * Hn + h) * Wn + px] = v + b_p[ch];
      else if (ch < 27)
        om[((size_t)ch * Hn + h) * Wn + px] = 1.f / (1.f + __expf(-(v + b_m[ch - 18])));
    }
  }
}

// ---------------------------------------------------------------------------
// Per (pixel, n): bilinear corner indices (flat pixel id, -1 => zero) and
// weights x modulation mask, matching the reference's clamping exactly.
// ---------------------------------------------------------------------------
__device__ __forceinline__ int flat_idx(int cx, int cy) {
  return (cx >= 1 && cx <= Hn && cy >= 1 && cy <= Wn) ? ((cx - 1) * Wn + (cy - 1)) : -1;
}

__global__ void coords_kernel(const float* __restrict__ offmask,
                              Rec* __restrict__ rec) {
  int gid = blockIdx.x * blockDim.x + threadIdx.x;   // exactly B*H*W
  int pw = gid & (Wn - 1);
  int ph = (gid >> 7) & (Hn - 1);
  int b  = gid >> 14;
  const float* om = offmask + (size_t)b * 27 * Hn * Wn + (size_t)ph * Wn + pw;
  Rec* rp = rec + (size_t)gid * Nk;
  const float hm = (float)(Hn + 1);   // padded-frame max coord (129)

#pragma unroll
  for (int n = 0; n < Nk; ++n) {
    float ox = om[(size_t)n * Hn * Wn];
    float oy = om[(size_t)(n + 9) * Hn * Wn];
    float m  = om[(size_t)(n + 18) * Hn * Wn];
    float pxf = (float)(ph + 1) + (float)(n / 3 - 1) + ox;
    float pyf = (float)(pw + 1) + (float)(n % 3 - 1) + oy;
    float fx = floorf(pxf), fy = floorf(pyf);
    float qltx = fminf(fmaxf(fx,       0.f), hm);
    float qlty = fminf(fmaxf(fy,       0.f), hm);
    float qrbx = fminf(fmaxf(fx + 1.f, 0.f), hm);
    float qrby = fminf(fmaxf(fy + 1.f, 0.f), hm);
    float pxc  = fminf(fmaxf(pxf,      0.f), hm);
    float pyc  = fminf(fmaxf(pyf,      0.f), hm);
    float glt = (1.f + (qltx - pxc)) * (1.f + (qlty - pyc));
    float grb = (1.f - (qrbx - pxc)) * (1.f - (qrby - pyc));
    float glb = (1.f + (qltx - pxc)) * (1.f - (qrby - pyc));
    float grt = (1.f - (qrbx - pxc)) * (1.f + (qlty - pyc));
    int x0 = (int)qltx, y0 = (int)qlty, x1 = (int)qrbx, y1 = (int)qrby;
    Rec r;
    r.idx[0] = flat_idx(x0, y0); r.wt[0] = glt * m;
    r.idx[1] = flat_idx(x1, y1); r.wt[1] = grb * m;
    r.idx[2] = flat_idx(x0, y1); r.wt[2] = glb * m;
    r.idx[3] = flat_idx(x1, y0); r.wt[3] = grt * m;
    rp[n] = r;
  }
}

// ---------------------------------------------------------------------------
// Fused bilinear-sample + final conv: GEMM M=64, K=576 (k = n*64+ci).
// One 16px x 64co tile per wave; records staged in LDS via TDM; NHWC gathers
// (4 x b128 per corner).  mode 0: ReLU -> NHWC y1; mode 1: +resid -> NCHW out.
// ---------------------------------------------------------------------------
__global__ void __launch_bounds__(128)
dcn_gemm_kernel(const float* __restrict__ xT,        // NHWC source
                const Rec* __restrict__ rec,
                const bf16_t* __restrict__ apack,    // MT=4
                const float* __restrict__ resid,     // NCHW (mode 1)
                float* __restrict__ out,
                int mode) {
  __shared__ Rec srec[4][16 * Nk];
  const int lane = threadIdx.x & 31;
  const int wv   = threadIdx.x >> 5;
  const int tile = blockIdx.x * 4 + wv;
  const int w0   = (tile & 7) * 16;
  const int h    = (tile >> 3) & (Hn - 1);
  const int b    = tile >> 10;
  const int px   = w0 + (lane & 15);
  const int ciL  = lane & 16;
  const size_t pixbase = ((size_t)b * Hn + h) * Wn + w0;

#if __has_builtin(__builtin_amdgcn_tensor_load_to_lds) && __has_builtin(__builtin_amdgcn_s_wait_tensorcnt)
  {
    // TDM: 1-D copy of this wave's 144 records (4608 B) into LDS.
    // D# group0: count=1, lds_addr, 57-bit global addr, type=2.
    // D# group1: data_size=8B, tensor_dim0=tile_dim0=576, dim1=1.
    unsigned long long ga = (unsigned long long)(uintptr_t)(const void*)(rec + pixbase * Nk);
    uint32_t lds_off = (uint32_t)(uintptr_t)(void*)(&srec[wv][0]);  // LDS aperture: addr[31:0]
    v4u g0 = { 1u, lds_off, (unsigned int)(ga & 0xffffffffu),
               (unsigned int)((ga >> 32) & 0x01ffffffu) | 0x80000000u };
    v8i g1 = { 0x30000, (int)(576u << 16), (int)(1u << 16), (int)(576u << 16),
               0, 576, 0, 0 };
    v4i gz = { 0, 0, 0, 0 };
#if defined(__clang_major__) && (__clang_major__ >= 23)
    v8i gz8 = { 0, 0, 0, 0, 0, 0, 0, 0 };
    __builtin_amdgcn_tensor_load_to_lds(g0, g1, gz, gz, gz8, 0);
#else
    __builtin_amdgcn_tensor_load_to_lds(g0, g1, gz, gz, 0);
#endif
    __builtin_amdgcn_s_wait_tensorcnt((short)0);
  }
#else
  for (int i = lane; i < 16 * Nk; i += 32) srec[wv][i] = rec[pixbase * Nk + i];
#endif
  __syncthreads();

  v8f acc[4] = { v8f{}, v8f{}, v8f{}, v8f{} };
  const float* xb = xT + (size_t)b * (Hn * Wn) * Cn;

  for (int t = 0; t < NCHUNK; ++t) {
    const int n   = t >> 1;
    const int ci0 = (t & 1) * 32 + ciL;
    Rec r = srec[wv][(lane & 15) * Nk + n];
    float wt0 = (r.idx[0] >= 0) ? r.wt[0] : 0.f;  size_t i0 = (size_t)max(r.idx[0], 0);
    float wt1 = (r.idx[1] >= 0) ? r.wt[1] : 0.f;  size_t i1 = (size_t)max(r.idx[1], 0);
    float wt2 = (r.idx[2] >= 0) ? r.wt[2] : 0.f;  size_t i2 = (size_t)max(r.idx[2], 0);
    float wt3 = (r.idx[3] >= 0) ? r.wt[3] : 0.f;  size_t i3 = (size_t)max(r.idx[3], 0);

    // 4 corners x 16 contiguous channels (NHWC) -> 16 x b128 loads.
    F16x c0, c1, c2, c3;
    {
      const v4f* q0 = (const v4f*)(xb + i0 * Cn + ci0);
      const v4f* q1 = (const v4f*)(xb + i1 * Cn + ci0);
      const v4f* q2 = (const v4f*)(xb + i2 * Cn + ci0);
      const v4f* q3 = (const v4f*)(xb + i3 * Cn + ci0);
#pragma unroll
      for (int u = 0; u < 4; ++u) { c0.f4[u] = q0[u]; c1.f4[u] = q1[u]; c2.f4[u] = q2[u]; c3.f4[u] = q3[u]; }
    }

    union { v16bf v; bf16_t e[16]; } bh, bl;
#pragma unroll
    for (int j = 0; j < 16; ++j) {
      float v = wt0 * c0.f[j] + wt1 * c1.f[j] + wt2 * c2.f[j] + wt3 * c3.f[j];
      split_bf16(v, bh.e[j], bl.e[j]);
    }

    const bf16_t* ap = apack + (((size_t)t * 4) * 32 + lane) * 32;
    __builtin_prefetch((const void*)(ap + 4096), 0, 0);   // next chunk's A frags
#pragma unroll
    for (int mt = 0; mt < 4; ++mt) {
      v16bf ah = *(const v16bf*)(ap + mt * 1024);
      v16bf al = *(const v16bf*)(ap + mt * 1024 + 16);
      acc[mt] = __builtin_amdgcn_wmma_f32_16x16x32_bf16(false, ah, false, bh.v, (short)0, acc[mt], false, false);
      acc[mt] = __builtin_amdgcn_wmma_f32_16x16x32_bf16(false, ah, false, bl.v, (short)0, acc[mt], false, false);
      acc[mt] = __builtin_amdgcn_wmma_f32_16x16x32_bf16(false, al, false, bh.v, (short)0, acc[mt], false, false);
    }
  }

  const int rsel = (lane & 16) ? 8 : 0;
  if (mode == 0) {
    // ReLU, NHWC output: 8 contiguous floats per M-tile per lane.
    float* op = out + (pixbase + (lane & 15)) * Cn + rsel;
#pragma unroll
    for (int mt = 0; mt < 4; ++mt) {
      v4f lo4 = { fmaxf(acc[mt][0], 0.f), fmaxf(acc[mt][1], 0.f),
                  fmaxf(acc[mt][2], 0.f), fmaxf(acc[mt][3], 0.f) };
      v4f hi4 = { fmaxf(acc[mt][4], 0.f), fmaxf(acc[mt][5], 0.f),
                  fmaxf(acc[mt][6], 0.f), fmaxf(acc[mt][7], 0.f) };
      *(v4f*)(op + mt * 16)     = lo4;
      *(v4f*)(op + mt * 16 + 4) = hi4;
    }
  } else {
    // Residual add, NCHW output.
#pragma unroll
    for (int mt = 0; mt < 4; ++mt) {
#pragma unroll
      for (int rr = 0; rr < 8; ++rr) {
        int co = mt * 16 + rr + rsel;
        size_t oidx = (((size_t)b * Cn + co) * Hn + h) * Wn + px;
        out[oidx] = acc[mt][rr] + resid[oidx];
      }
    }
  }
}

// ---------------------------------------------------------------------------
extern "C" void kernel_launch(void* const* d_in, const int* in_sizes, int n_in,
                              void* d_out, int out_size, void* d_ws, size_t ws_size,
                              hipStream_t stream) {
  (void)in_sizes; (void)n_in; (void)out_size; (void)ws_size;
  const float* x    = (const float*)d_in[0];
  const float* w_p1 = (const float*)d_in[1];
  const float* b_p1 = (const float*)d_in[2];
  const float* w_m1 = (const float*)d_in[3];
  const float* b_m1 = (const float*)d_in[4];
  const float* w_c1 = (const float*)d_in[5];
  const float* w_p2 = (const float*)d_in[6];
  const float* b_p2 = (const float*)d_in[7];
  const float* w_m2 = (const float*)d_in[8];
  const float* b_m2 = (const float*)d_in[9];
  const float* w_c2 = (const float*)d_in[10];
  float* out = (float*)d_out;

  char*  ws = (char*)d_ws;
  size_t o  = 0;
  auto take = [&](size_t bytes) -> void* {
    void* p = ws + o;
    o += (bytes + 255) & ~(size_t)255;
    return p;
  };
  bf16_t* apOM1   = (bf16_t*)take((size_t)NCHUNK * 2 * 32 * 32 * sizeof(bf16_t));
  bf16_t* apC1    = (bf16_t*)take((size_t)NCHUNK * 4 * 32 * 32 * sizeof(bf16_t));
  bf16_t* apOM2   = (bf16_t*)take((size_t)NCHUNK * 2 * 32 * 32 * sizeof(bf16_t));
  bf16_t* apC2    = (bf16_t*)take((size_t)NCHUNK * 4 * 32 * 32 * sizeof(bf16_t));
  float*  offmask = (float*)take((size_t)Bn * 27 * Hn * Wn * sizeof(float));
  Rec*    rec     = (Rec*)take((size_t)Bn * Hn * Wn * Nk * sizeof(Rec));
  float*  xT      = (float*)take((size_t)Bn * Hn * Wn * Cn * sizeof(float)); // NHWC x
  float*  y1      = (float*)take((size_t)Bn * Hn * Wn * Cn * sizeof(float)); // NHWC y1

  // Weight repack + input transpose (deterministic, every call).
  pack_a_kernel<<<72, 256, 0, stream>>>(w_p1, 18, w_m1, 9, apOM1, 2);
  pack_a_kernel<<<144, 256, 0, stream>>>(w_c1, 64, w_c1, 0, apC1, 4);
  pack_a_kernel<<<72, 256, 0, stream>>>(w_p2, 18, w_m2, 9, apOM2, 2);
  pack_a_kernel<<<144, 256, 0, stream>>>(w_c2, 64, w_c2, 0, apC2, 4);
  nchw_to_nhwc_kernel<<<1024, 256, 0, stream>>>(x, xT);

  // Layer 1: xT -> y1 (NHWC, ReLU)
  conv_offmask_kernel<<<1024, 128, 0, stream>>>(xT, apOM1, b_p1, b_m1, offmask);
  coords_kernel<<<256, 256, 0, stream>>>(offmask, rec);
  dcn_gemm_kernel<<<1024, 128, 0, stream>>>(xT, rec, apC1, x, y1, 0);

  // Layer 2: y1 -> out (NCHW, + x residual)
  conv_offmask_kernel<<<1024, 128, 0, stream>>>(y1, apOM2, b_p2, b_m2, offmask);
  coords_kernel<<<256, 256, 0, stream>>>(offmask, rec);
  dcn_gemm_kernel<<<1024, 128, 0, stream>>>(y1, rec, apC2, x, out, 1);
}
